// Model_48971217109711
// MI455X (gfx1250) — matrix operand order repaired and checked by reading, not compiled, not run on hardware
//
#include <hip/hip_runtime.h>
#include <hip/hip_bf16.h>

typedef __attribute__((ext_vector_type(16))) _Float16 v16h;
typedef __attribute__((ext_vector_type(8)))  _Float16 v8h;
typedef __attribute__((ext_vector_type(8)))  float    v8f;

#define TB 2
#define TC 3
#define TS 512
#define TD 512
#define THN 8
#define THD 64
#define TFF 2048
#define TDEPTH 6
#define TNC 10
#define TN 1537            /* C*S + 1 */
#define TM (TB * TN)       /* 3074 rows */
#define TCS (TC * TS)

// ---------------------------------------------------------------- WMMA helpers
__device__ __forceinline__ v8f wmma16(v16h a, v16h b, v8f c) {
  return __builtin_amdgcn_wmma_f32_16x16x32_f16(false, a, false, b, (short)0, c,
                                                false, false);
}
// A-matrix 16x32 f16 fragment: lane holds row M=lane&15, 16 contiguous K
// halves starting at 16*(lane>=16). p -> first element.
__device__ __forceinline__ v16h frag_a16(const _Float16* p) {
  v8h lo = *(const v8h*)p;
  v8h hi = *(const v8h*)(p + 8);
  v16h r;
#pragma unroll
  for (int i = 0; i < 8; i++) { r[i] = lo[i]; r[i + 8] = hi[i]; }
  return r;
}
// B-matrix 32x16 f16 fragment: lane holds column N=lane&15, 16 contiguous K
// halves starting at 16*(lane>=16). Requires 32B alignment.
__device__ __forceinline__ v16h frag_b16(const _Float16* p) {
  return *(const v16h*)p;
}
__device__ __forceinline__ float gelu_f(float x) {
  float x3 = x * x * x;
  return 0.5f * x * (1.f + tanhf(0.7978845608028654f * (x + 0.044715f * x3)));
}

// -------------------------------------------------- weight convert + transpose
// W: fp32 [K,N] row-major -> Wt: f16 [N,K] (so B-fragments load contiguously)
__global__ void cvt_transpose_kernel(const float* __restrict__ W,
                                     _Float16* __restrict__ Wt, int K, int N) {
  size_t t = (size_t)blockIdx.x * 256 + threadIdx.x;
  if (t >= (size_t)K * N) return;
  int nn = (int)(t % N);
  int kk = (int)(t / N);
  Wt[(size_t)nn * K + kk] = (_Float16)W[t];
}

// ------------------------------------------------------------------- embedding
__global__ void embed_kernel(const float* __restrict__ x,
                             const float* __restrict__ emb,
                             const float* __restrict__ cls,
                             float* __restrict__ h) {
  size_t t = (size_t)blockIdx.x * 256 + threadIdx.x;
  if (t >= (size_t)TM * TD) return;
  int d = (int)(t % TD);
  size_t rem = t / TD;
  int p = (int)(rem % TN);
  int b = (int)(rem / TN);
  float val;
  if (p == 0) {
    val = cls[d];
  } else {
    int tok = (int)(x[(size_t)b * TCS + (p - 1)] * 1000.0f);
    val = emb[(size_t)tok * TD + d];
  }
  h[t] = val;
}

// ------------------------------------------------------------------- layernorm
__global__ __launch_bounds__(256) void ln_kernel(const float* __restrict__ x,
                                                 const float* __restrict__ s,
                                                 const float* __restrict__ bb,
                                                 _Float16* __restrict__ outH,
                                                 float* __restrict__ outF) {
  int row = blockIdx.x;
  const float* xr = x + (size_t)row * TD;
  float sum = 0.f, sq = 0.f;
#pragma unroll
  for (int c = threadIdx.x; c < TD; c += 256) {
    float v = xr[c]; sum += v; sq += v * v;
  }
#pragma unroll
  for (int m = 16; m; m >>= 1) {
    sum += __shfl_xor(sum, m, 32);
    sq  += __shfl_xor(sq,  m, 32);
  }
  __shared__ float s1[8], s2[8];
  int wave = threadIdx.x >> 5, lane = threadIdx.x & 31;
  if (lane == 0) { s1[wave] = sum; s2[wave] = sq; }
  __syncthreads();
  sum = 0.f; sq = 0.f;
#pragma unroll
  for (int i = 0; i < 8; i++) { sum += s1[i]; sq += s2[i]; }
  float mean = sum * (1.0f / TD);
  float var  = sq * (1.0f / TD) - mean * mean;
  float rstd = rsqrtf(var + 1e-5f);
#pragma unroll
  for (int c = threadIdx.x; c < TD; c += 256) {
    float y = (xr[c] - mean) * rstd * s[c] + bb[c];
    if (outH) outH[(size_t)row * TD + c] = (_Float16)y;
    if (outF) outF[(size_t)row * TD + c] = y;
  }
}

// ------------------------------------------------------------------- WMMA GEMM
// out[M,N] = act(A[M,K] @ Wt^T + bias) + res ; Wt is f16 [N,K].
// Each wave: 32x64 output tile (2 A-frags x 4 B-frags -> 8 wmma per K-step).
__global__ __launch_bounds__(256) void gemm_kernel(
    const _Float16* __restrict__ A, const _Float16* __restrict__ Bt,
    const float* __restrict__ bias, const float* __restrict__ res,
    float* __restrict__ outF, _Float16* __restrict__ outH,
    int M, int N, int K, int act) {
  const int lane = threadIdx.x & 31;
  const int wave = threadIdx.x >> 5;
  const int hlf  = lane >> 4;
  const int lc   = lane & 15;
  const int m0 = blockIdx.x * 256 + wave * 32;
  const int n0 = blockIdx.y * 64;
  if (m0 >= M) return;  // uniform per wave: EXEC stays all-ones for WMMA
  v8f acc[2][4];
#pragma unroll
  for (int i = 0; i < 2; i++)
#pragma unroll
    for (int t = 0; t < 4; t++)
#pragma unroll
      for (int r = 0; r < 8; r++) acc[i][t][r] = 0.f;
  const int ra0 = min(m0 + lc,      M - 1);
  const int ra1 = min(m0 + 16 + lc, M - 1);
  const int koffA = hlf << 4;
  const int koffB = hlf << 4;
  for (int kb = 0; kb < K; kb += 32) {
    v16h a0 = frag_a16(A + (size_t)ra0 * K + kb + koffA);
    v16h a1 = frag_a16(A + (size_t)ra1 * K + kb + koffA);
#pragma unroll
    for (int t = 0; t < 4; t++) {
      v16h b = frag_b16(Bt + (size_t)(n0 + t * 16 + lc) * K + kb + koffB);
      acc[0][t] = wmma16(a0, b, acc[0][t]);
      acc[1][t] = wmma16(a1, b, acc[1][t]);
    }
  }
#pragma unroll
  for (int i = 0; i < 2; i++) {
#pragma unroll
    for (int t = 0; t < 4; t++) {
      int nn = n0 + t * 16 + lc;
      float bv = bias ? bias[nn] : 0.f;
#pragma unroll
      for (int r = 0; r < 8; r++) {
        int mm = m0 + i * 16 + hlf * 8 + r;
        if (mm < M) {
          float val = acc[i][t][r] + bv;
          if (act) val = gelu_f(val);
          if (res)  val += res[(size_t)mm * N + nn];
          if (outF) outF[(size_t)mm * N + nn] = val;
          if (outH) outH[(size_t)mm * N + nn] = (_Float16)val;
        }
      }
    }
  }
}

// ------------------------------------------------------------- flash attention
// One wave per (batch, head, 16-query tile). q/k/v/o: f16 [B*n, D],
// head hh occupies columns hh*64..hh*64+63.
#define VT_LD 48  /* halves; 96B row stride -> 16B-aligned fragment loads */
__global__ __launch_bounds__(32) void attn_kernel(
    const _Float16* __restrict__ q, const _Float16* __restrict__ k,
    const _Float16* __restrict__ v, _Float16* __restrict__ o) {
  __shared__ __align__(32) _Float16 Pl[16 * 32];
  __shared__ __align__(32) _Float16 Vt[64 * VT_LD];
  const int lane = threadIdx.x;
  const int hlf  = lane >> 4;
  const int lc   = lane & 15;
  const int m0 = blockIdx.x * 16;
  const int hh = blockIdx.y;
  const int bb = blockIdx.z;
  const _Float16* qb = q + (size_t)bb * TN * TD + hh * THD;
  const _Float16* kb = k + (size_t)bb * TN * TD + hh * THD;
  const _Float16* vb = v + (size_t)bb * TN * TD + hh * THD;
  const float slope = (hh < 4) ? exp2f(-2.0f * (float)(hh + 1)) : 0.f;
  const float scale = 0.125f;  // HD^-0.5

  const int ra = min(m0 + lc, TN - 1);
  v16h qa0 = frag_a16(qb + (size_t)ra * TD + 0  + (hlf << 4));
  v16h qa1 = frag_a16(qb + (size_t)ra * TD + 32 + (hlf << 4));

  v8f accO[4];
  float mrow[8], lrow[8];
#pragma unroll
  for (int t = 0; t < 4; t++)
#pragma unroll
    for (int r = 0; r < 8; r++) accO[t][r] = 0.f;
#pragma unroll
  for (int r = 0; r < 8; r++) { mrow[r] = -INFINITY; lrow[r] = 0.f; }

  for (int j0 = 0; j0 < TN; j0 += 32) {
    __syncthreads();
    // stage V chunk transposed: Vt[d][jj] = V[j0+jj][d]
    for (int t = lane; t < 32 * 64; t += 32) {
      int jj = t >> 6;
      int d  = t & 63;
      int jr = min(j0 + jj, TN - 1);
      Vt[d * VT_LD + jj] = vb[(size_t)jr * TD + d];
    }
    __syncthreads();

    // logits: two 16x16 tiles over keys j0..j0+31 (K-dim = head dim, 2 steps)
    float lv[2][8];
#pragma unroll
    for (int t = 0; t < 2; t++) {
      int jr = min(j0 + t * 16 + lc, TN - 1);
      v16h b0 = frag_b16(kb + (size_t)jr * TD + 0  + (hlf << 4));
      v16h b1 = frag_b16(kb + (size_t)jr * TD + 32 + (hlf << 4));
      v8f c;
#pragma unroll
      for (int r = 0; r < 8; r++) c[r] = 0.f;
      c = wmma16(qa0, b0, c);
      c = wmma16(qa1, b1, c);
      int j = j0 + t * 16 + lc;  // this lane's key column
#pragma unroll
      for (int r = 0; r < 8; r++) {
        int i = m0 + hlf * 8 + r;  // query row
        float val = c[r] * scale - slope * fabsf((float)(i - j));
        lv[t][r] = (j < TN) ? val : -INFINITY;
      }
    }

    // online softmax per row (rows live across the 16 lanes of each half)
#pragma unroll
    for (int r = 0; r < 8; r++) {
      float mx = fmaxf(lv[0][r], lv[1][r]);
      mx = fmaxf(mx, __shfl_xor(mx, 1, 32));
      mx = fmaxf(mx, __shfl_xor(mx, 2, 32));
      mx = fmaxf(mx, __shfl_xor(mx, 4, 32));
      mx = fmaxf(mx, __shfl_xor(mx, 8, 32));
      float nm = fmaxf(mrow[r], mx);
      float corr = __expf(mrow[r] - nm);
      mrow[r] = nm;
      float p0 = __expf(lv[0][r] - nm);
      float p1 = __expf(lv[1][r] - nm);
      float ps = p0 + p1;
      ps += __shfl_xor(ps, 1, 32);
      ps += __shfl_xor(ps, 2, 32);
      ps += __shfl_xor(ps, 4, 32);
      ps += __shfl_xor(ps, 8, 32);
      lrow[r] = lrow[r] * corr + ps;
#pragma unroll
      for (int t = 0; t < 4; t++) accO[t][r] *= corr;
      int ml = hlf * 8 + r;
      Pl[ml * 32 + lc]      = (_Float16)p0;
      Pl[ml * 32 + 16 + lc] = (_Float16)p1;
    }
    __syncthreads();

    // O += P @ V : A = P(16x32) from LDS, B = Vt fragments, 4 d-tiles
    v16h pa = frag_a16(Pl + lc * 32 + (hlf << 4));
#pragma unroll
    for (int t = 0; t < 4; t++) {
      v16h vbf = frag_b16(Vt + (size_t)(t * 16 + lc) * VT_LD + (hlf << 4));
      accO[t] = wmma16(pa, vbf, accO[t]);
    }
  }

  // normalize + store f16 output in [B*n, D] layout
#pragma unroll
  for (int t = 0; t < 4; t++) {
#pragma unroll
    for (int r = 0; r < 8; r++) {
      int mm = m0 + hlf * 8 + r;
      if (mm < TN) {
        float val = accO[t][r] / lrow[r];
        o[((size_t)bb * TN + mm) * TD + hh * THD + t * 16 + lc] = (_Float16)val;
      }
    }
  }
}

// ------------------------------------------------------------------ classifier
__global__ __launch_bounds__(32) void cls_kernel(const float* __restrict__ hf,
                                                 const float* __restrict__ wc,
                                                 const float* __restrict__ bc,
                                                 float* __restrict__ y) {
  int b = blockIdx.x;
  int lane = threadIdx.x;
  float logit = -INFINITY;
  if (lane < TNC) {
    float acc = bc[lane];
    const float* hr = hf + (size_t)b * TN * TD;  // token 0 row
    for (int d = 0; d < TD; d++) acc += hr[d] * wc[(size_t)d * TNC + lane];
    logit = acc;
  }
  float mx = logit;
#pragma unroll
  for (int m = 16; m; m >>= 1) mx = fmaxf(mx, __shfl_xor(mx, m, 32));
  float e = (lane < TNC) ? __expf(logit - mx) : 0.f;
  float s = e;
#pragma unroll
  for (int m = 16; m; m >>= 1) s += __shfl_xor(s, m, 32);
  if (lane < TNC) y[(size_t)b * TNC + lane] = e / s;
}

// -------------------------------------------------------------------- launcher
extern "C" void kernel_launch(void* const* d_in, const int* in_sizes, int n_in,
                              void* d_out, int out_size, void* d_ws,
                              size_t ws_size, hipStream_t stream) {
  (void)in_sizes; (void)n_in; (void)out_size; (void)ws_size;
  const float* x      = (const float*)d_in[0];
  const float* embT   = (const float*)d_in[1];
  const float* clsT   = (const float*)d_in[2];
  const float* ln1_s  = (const float*)d_in[3];
  const float* ln1_b  = (const float*)d_in[4];
  const float* wq     = (const float*)d_in[5];
  const float* wk     = (const float*)d_in[6];
  const float* wv     = (const float*)d_in[7];
  const float* wo     = (const float*)d_in[8];
  const float* bo     = (const float*)d_in[9];
  const float* ln2_s  = (const float*)d_in[10];
  const float* ln2_b  = (const float*)d_in[11];
  const float* w1     = (const float*)d_in[12];
  const float* b1     = (const float*)d_in[13];
  const float* w2     = (const float*)d_in[14];
  const float* b2     = (const float*)d_in[15];
  const float* lnf_s  = (const float*)d_in[16];
  const float* lnf_b  = (const float*)d_in[17];
  const float* wc     = (const float*)d_in[18];
  const float* bc     = (const float*)d_in[19];

  char* base = (char*)d_ws;
  size_t off = 0;
  auto take = [&](size_t bytes) -> void* {
    void* p = base + off;
    off = (off + bytes + 255) & ~(size_t)255;
    return p;
  };
  _Float16* wtq = (_Float16*)take((size_t)TDEPTH * TD * TD * 2);
  _Float16* wtk = (_Float16*)take((size_t)TDEPTH * TD * TD * 2);
  _Float16* wtv = (_Float16*)take((size_t)TDEPTH * TD * TD * 2);
  _Float16* wto = (_Float16*)take((size_t)TDEPTH * TD * TD * 2);
  _Float16* wt1 = (_Float16*)take((size_t)TDEPTH * TD * TFF * 2);
  _Float16* wt2 = (_Float16*)take((size_t)TDEPTH * TFF * TD * 2);
  float*    h    = (float*)   take((size_t)TM * TD * 4);
  _Float16* xn_h = (_Float16*)take((size_t)TM * TD * 2);
  _Float16* q_h  = (_Float16*)take((size_t)TM * TD * 2);
  _Float16* k_h  = (_Float16*)take((size_t)TM * TD * 2);
  _Float16* v_h  = (_Float16*)take((size_t)TM * TD * 2);
  _Float16* o_h  = (_Float16*)take((size_t)TM * TD * 2);
  _Float16* ff_h = (_Float16*)take((size_t)TM * TFF * 2);

  // 1) convert + transpose all weights to f16 [N,K]
  const int gDD = (TD * TD + 255) / 256;
  const int gDF = (TD * TFF + 255) / 256;
  for (int l = 0; l < TDEPTH; l++) {
    cvt_transpose_kernel<<<gDD, 256, 0, stream>>>(wq + (size_t)l * TD * TD,  wtq + (size_t)l * TD * TD,  TD, TD);
    cvt_transpose_kernel<<<gDD, 256, 0, stream>>>(wk + (size_t)l * TD * TD,  wtk + (size_t)l * TD * TD,  TD, TD);
    cvt_transpose_kernel<<<gDD, 256, 0, stream>>>(wv + (size_t)l * TD * TD,  wtv + (size_t)l * TD * TD,  TD, TD);
    cvt_transpose_kernel<<<gDD, 256, 0, stream>>>(wo + (size_t)l * TD * TD,  wto + (size_t)l * TD * TD,  TD, TD);
    cvt_transpose_kernel<<<gDF, 256, 0, stream>>>(w1 + (size_t)l * TD * TFF, wt1 + (size_t)l * TD * TFF, TD, TFF);
    cvt_transpose_kernel<<<gDF, 256, 0, stream>>>(w2 + (size_t)l * TFF * TD, wt2 + (size_t)l * TFF * TD, TFF, TD);
  }

  // 2) embedding (+CLS)
  embed_kernel<<<((size_t)TM * TD + 255) / 256, 256, 0, stream>>>(x, embT, clsT, h);

  const dim3 gN512((TM + 255) / 256, TD / 64);
  const dim3 gNFF((TM + 255) / 256, TFF / 64);
  const dim3 gAttn((TN + 15) / 16, THN, TB);

  for (int l = 0; l < TDEPTH; l++) {
    // pre-norm attention
    ln_kernel<<<TM, 256, 0, stream>>>(h, ln1_s + (size_t)l * TD, ln1_b + (size_t)l * TD, xn_h, nullptr);
    gemm_kernel<<<gN512, 256, 0, stream>>>(xn_h, wtq + (size_t)l * TD * TD, nullptr, nullptr, nullptr, q_h, TM, TD, TD, 0);
    gemm_kernel<<<gN512, 256, 0, stream>>>(xn_h, wtk + (size_t)l * TD * TD, nullptr, nullptr, nullptr, k_h, TM, TD, TD, 0);
    gemm_kernel<<<gN512, 256, 0, stream>>>(xn_h, wtv + (size_t)l * TD * TD, nullptr, nullptr, nullptr, v_h, TM, TD, TD, 0);
    attn_kernel<<<gAttn, 32, 0, stream>>>(q_h, k_h, v_h, o_h);
    gemm_kernel<<<gN512, 256, 0, stream>>>(o_h, wto + (size_t)l * TD * TD, bo + (size_t)l * TD, h, h, nullptr, TM, TD, TD, 0);
    // pre-norm FFN
    ln_kernel<<<TM, 256, 0, stream>>>(h, ln2_s + (size_t)l * TD, ln2_b + (size_t)l * TD, xn_h, nullptr);
    gemm_kernel<<<gNFF, 256, 0, stream>>>(xn_h, wt1 + (size_t)l * TD * TFF, b1 + (size_t)l * TFF, nullptr, nullptr, ff_h, TM, TFF, TD, 1);
    gemm_kernel<<<gN512, 256, 0, stream>>>(ff_h, wt2 + (size_t)l * TFF * TD, b2 + (size_t)l * TD, h, h, nullptr, TM, TD, TFF, 0);
  }

  // final layernorm -> d_out h region; classifier softmax -> d_out y region
  float* y    = (float*)d_out;
  float* hout = y + TB * TNC;
  ln_kernel<<<TM, 256, 0, stream>>>(h, lnf_s, lnf_b, nullptr, hout);
  cls_kernel<<<TB, 32, 0, stream>>>(hout, wc, bc, y);
}
